// LocalNorm2d_82282983457438
// MI455X (gfx1250) — matrix operand order repaired and checked by reading, not compiled, not run on hardware
//
#include <hip/hip_runtime.h>

// LocalNorm2d: out = clip((x - boxmean(x)) / (sqrt(|boxmean(x^2) - boxmean(x)^2|) + 1e-10), -6, 6)
// Box = 32x32, stride 1, reflect padding 16. Input (32,3,512,512) f32.
//
// Strategy: one workgroup per 64x64 output tile.
//   - Load 95x95 reflect-padded input tile to LDS (f32), zero-padded to 112 rows
//     so the WMMA feed loop needs no bounds checks (no EXEC-predicated DS loads).
//   - Vertical 32-row sliding sums (and sum of squares) computed as banded-ones
//     matmuls on the matrix cores: V_WMMA_F32_16X16X32_F16, f32 accumulate.
//   - Horizontal 32-col sliding sums as O(1)/output f32 VALU scans + normalize.
//   - Stage results in LDS, fully coalesced global store.

typedef __attribute__((ext_vector_type(16))) _Float16 v16h;
typedef __attribute__((ext_vector_type(8)))  float    v8f;

#define HH   512
#define WW   512
#define KS   32
#define PD   16
#define TILE 64
#define XROWS 95           // TILE + KS - 1 (valid rows)
#define XPAD  112          // rows incl. zero padding: max WMMA row = 48+63 = 111
#define XSTR  97           // padded LDS stride for x tile (bank-conflict free)
#define CSTR  98           // padded LDS stride for colsum tiles
#define OSTR  65           // padded LDS stride for output staging

__global__ __launch_bounds__(256)
void localnorm2d_kernel(const float* __restrict__ x, float* __restrict__ out) {
    extern __shared__ float smem[];
    float* xs  = smem;                         // [112][97] input tile (f32), rows 95..111 zero
    float* cs  = xs  + XPAD * XSTR;            // [64][98]  vertical sliding sum
    float* cs2 = cs  + TILE * CSTR;            // [64][98]  vertical sliding sum of squares
    float* ot  = cs2 + TILE * CSTR;            // [64][65]  output staging

    const int tid  = threadIdx.x;
    const int lane = tid & 31;
    const int wave = tid >> 5;
    const int img  = blockIdx.z;               // batch*channel plane, 0..95
    const int h0   = blockIdx.y * TILE;
    const int w0   = blockIdx.x * TILE;
    const float* xi = x   + (size_t)img * (HH * WW);
    float*       oi = out + (size_t)img * (HH * WW);

    // ---------------- Stage 1: load reflect-padded tile (wave-per-row) ----------------
    // Reflected column indices for this lane's three column slots (0..95).
    int gcol[3];
#pragma unroll
    for (int k = 0; k < 3; ++k) {
        int c  = lane + 32 * k;
        int gc = w0 - PD + c;
        gc = (gc < 0) ? -gc : ((gc >= WW) ? (2 * WW - 2 - gc) : gc);
        gcol[k] = gc;
    }
    for (int r = wave; r < XPAD; r += 8) {
        if (r < XROWS) {
            int gr = h0 - PD + r;
            gr = (gr < 0) ? -gr : ((gr >= HH) ? (2 * HH - 2 - gr) : gr);
            const float* rowp = xi + (size_t)gr * WW;
#pragma unroll
            for (int k = 0; k < 3; ++k) {
                int c = lane + 32 * k;
                float v = (c < XROWS) ? rowp[gcol[k]] : 0.0f;   // only lane31@k=2 masked
                xs[r * XSTR + c] = v;
            }
        } else {
#pragma unroll
            for (int k = 0; k < 3; ++k) {
                int c = lane + 32 * k;
                if (c < 96) xs[r * XSTR + c] = 0.0f;
            }
        }
    }
    __syncthreads();

    // ---------------- Stage 2: vertical 32-row sliding sums via WMMA ----------------
    // colsum[r0+m][c0+n] = sum_{j=m..m+31} xs[r0+j][c0+n]  -> D = A(16x32 banded) x B(32x16 data)
    const int m  = lane & 15;          // A-matrix row / B,C,D column index per lane
    const int hi = (lane >> 4) & 1;    // which half of the wave

    // Banded-ones A matrices for K-chunk 0 (j=0..31) and chunk 1 (j=32..63).
    // A layout (16x32 f16): lane m holds K = 2v + hi*8 (v<4) or 16 + 2(v-4) + hi*8.
    v16h A0, A1;
#pragma unroll
    for (int v = 0; v < 8; ++v) {
        int K = (v < 4) ? (2 * v + hi * 8) : (16 + 2 * (v - 4) + hi * 8);
#pragma unroll
        for (int h = 0; h < 2; ++h) {
            int jj0 = K + h;           // chunk 0: j index
            int jj1 = 32 + K + h;      // chunk 1
            _Float16 w0v = (_Float16)((jj0 >= m && jj0 <= m + 31) ? 1.0f : 0.0f);
            _Float16 w1v = (_Float16)((jj1 >= m && jj1 <= m + 31) ? 1.0f : 0.0f);
            if (h == 0) { A0[2 * v] = w0v;     A1[2 * v] = w1v; }
            else        { A0[2 * v + 1] = w0v; A1[2 * v + 1] = w1v; }
        }
    }

    // 24 blocks (4 row-blocks x 6 col-blocks) over 8 waves, 3 each — fully uniform.
#pragma unroll
    for (int bi = 0; bi < 3; ++bi) {
        int blk = wave + 8 * bi;
        int r0  = (blk / 6) * 16;
        int c0  = (blk % 6) * 16;

        v8f acc  = {0.f, 0.f, 0.f, 0.f, 0.f, 0.f, 0.f, 0.f};
        v8f acc2 = {0.f, 0.f, 0.f, 0.f, 0.f, 0.f, 0.f, 0.f};

#pragma unroll
        for (int ch = 0; ch < 2; ++ch) {
            // B layout (32x16 f16): lane n=m holds rows K = 2v + hi*8 (v<4) or 16 + 2(v-4) + hi*8 (+ch*32), col c0+n.
            v16h B, B2;
#pragma unroll
            for (int v = 0; v < 8; ++v) {
                int row0 = r0 + ch * 32 + 2 * v + hi * 8 + ((v >= 4) ? 8 : 0);   // <= 111 < XPAD, unconditional
                float f0 = xs[row0 * XSTR + c0 + m];
                float f1 = xs[(row0 + 1) * XSTR + c0 + m];
                B [2 * v]     = (_Float16)f0;
                B [2 * v + 1] = (_Float16)f1;
                B2[2 * v]     = (_Float16)(f0 * f0);
                B2[2 * v + 1] = (_Float16)(f1 * f1);
            }
            v16h Ach = (ch == 0) ? A0 : A1;
            acc  = __builtin_amdgcn_wmma_f32_16x16x32_f16(false, Ach, false, B,
                                                          (short)0, acc,  false, false);
            acc2 = __builtin_amdgcn_wmma_f32_16x16x32_f16(false, Ach, false, B2,
                                                          (short)0, acc2, false, false);
        }

        // D layout (16x16 f32): VGPR v holds M = v + 8*hi, column c0+m.
#pragma unroll
        for (int v = 0; v < 8; ++v) {
            int row = r0 + v + hi * 8;
            cs [row * CSTR + c0 + m] = acc[v];
            cs2[row * CSTR + c0 + m] = acc2[v];
        }
    }
    __syncthreads();

    // ---------------- Stage 3: horizontal sliding window + normalize ----------------
    {
        const int r  = tid >> 2;       // output row 0..63
        const int s  = tid & 3;        // column segment
        const int wb = 16 * s;
        float ssum = 0.0f, ssq = 0.0f;
#pragma unroll
        for (int j = 0; j < KS; ++j) {
            ssum += cs [r * CSTR + wb + j];
            ssq  += cs2[r * CSTR + wb + j];
        }
        const float inv = 1.0f / (float)(KS * KS);
#pragma unroll
        for (int i = 0; i < 16; ++i) {
            int w = wb + i;
            float mean = ssum * inv;
            float msq  = ssq  * inv;
            float sd   = sqrtf(fabsf(msq - mean * mean));
            float xv   = xs[(r + PD) * XSTR + (w + PD)];
            float res  = (xv - mean) / (sd + 1e-10f);
            res = fminf(6.0f, fmaxf(-6.0f, res));
            ot[r * OSTR + w] = res;
            // slide window: drop col w, add col w+32 (col 95 is defined zero)
            ssum += cs [r * CSTR + w + KS] - cs [r * CSTR + w];
            ssq  += cs2[r * CSTR + w + KS] - cs2[r * CSTR + w];
        }
    }
    __syncthreads();

    // ---------------- Stage 4: coalesced store ----------------
#pragma unroll
    for (int i = 0; i < 16; ++i) {
        int e = tid + 256 * i;
        int r = e >> 6;
        int c = e & 63;
        oi[(size_t)(h0 + r) * WW + (w0 + c)] = ot[r * OSTR + c];
    }
}

extern "C" void kernel_launch(void* const* d_in, const int* in_sizes, int n_in,
                              void* d_out, int out_size, void* d_ws, size_t ws_size,
                              hipStream_t stream) {
    (void)in_sizes; (void)n_in; (void)out_size; (void)d_ws; (void)ws_size;
    const float* x = (const float*)d_in[0];
    float* out = (float*)d_out;

    dim3 grid(WW / TILE, HH / TILE, 32 * 3);   // 8 x 8 x 96 tiles
    dim3 block(256);
    size_t smem = (size_t)(XPAD * XSTR + 2 * TILE * CSTR + TILE * OSTR) * sizeof(float);
    localnorm2d_kernel<<<grid, block, smem, stream>>>(x, out);
}
